// SigTKANDense_55413668053533
// MI455X (gfx1250) — matrix operand order repaired and checked by reading, not compiled, not run on hardware
//
#include <hip/hip_runtime.h>

// ---------------------------------------------------------------------------
// SigTKANDense for MI455X (gfx1250, wave32, WMMA).
//  - Signature level-2 rewritten as GEMM (prefix sums telescope; no scan).
//  - All GEMMs: v_wmma_f32_16x16x32_bf16, f32 accumulate.
//  - Weights pre-transposed to bf16 once (~10 MB; L2-resident, 192 MB L2).
//  - stk_final stages x via GLOBAL_LOAD_ASYNC_TO_LDS_B128 (ASYNCcnt path),
//    double-buffers the LDS A-panel (1 barrier/K-chunk).
//  - Long-K GEMMs use split-K + f32 global atomics for occupancy.
// ---------------------------------------------------------------------------

typedef __attribute__((ext_vector_type(16))) __bf16 bf16x16;
typedef __attribute__((ext_vector_type(8)))  __bf16 bf16x8;
typedef __attribute__((ext_vector_type(8)))  float  f32x8;

union FragBF { bf16x16 v; bf16x8 h[2]; };

__device__ inline f32x8 zero8() {
  f32x8 z = {0.f,0.f,0.f,0.f,0.f,0.f,0.f,0.f};
  return z;
}

__device__ inline __bf16 f2bf(float f) {
  union { float f; unsigned u; } in; in.f = f;
  unsigned r = (in.u + 0x7FFFu + ((in.u >> 16) & 1u)) >> 16;   // RNE
  union { unsigned short u; __bf16 b; } o; o.u = (unsigned short)r;
  return o.b;
}

__device__ inline float siluf(float v) { return v / (1.f + __expf(-v)); }

// Cox-de Boor, order 3, 12 uniform knots g[k] = (k-3)*0.4 - 1  -> 8 bases.
__device__ inline void bspline8(float x, float* out) {
  const float h = 0.4f;
  float b0[11];
#pragma unroll
  for (int k = 0; k < 11; ++k) {
    float gk = -1.0f + (k - 3) * h;
    b0[k] = (x >= gk && x < gk + h) ? 1.0f : 0.0f;
  }
  float b1[10];
#pragma unroll
  for (int k = 0; k < 10; ++k) {
    float gk = -1.0f + (k - 3) * h;
    b1[k] = (x - gk) * (1.f/(1.f*h)) * b0[k] + ((gk + 2*h) - x) * (1.f/(1.f*h)) * b0[k+1];
  }
  float b2[9];
#pragma unroll
  for (int k = 0; k < 9; ++k) {
    float gk = -1.0f + (k - 3) * h;
    b2[k] = (x - gk) * (1.f/(2.f*h)) * b1[k] + ((gk + 3*h) - x) * (1.f/(2.f*h)) * b1[k+1];
  }
#pragma unroll
  for (int k = 0; k < 8; ++k) {
    float gk = -1.0f + (k - 3) * h;
    out[k] = (x - gk) * (1.f/(3.f*h)) * b2[k] + ((gk + 4*h) - x) * (1.f/(3.f*h)) * b2[k+1];
  }
}

// --- Weight convert: [base;spline] (K x 128, f32) -> Wt (128 x K, bf16) -----
__global__ void stk_wconv(const float* __restrict__ base, const float* __restrict__ spl,
                          __bf16* __restrict__ out, int Kb, int Ks) {
  int K = Kb + Ks;
  int idx = blockIdx.x * blockDim.x + threadIdx.x;
  if (idx >= 128 * K) return;
  int n = idx / K, k = idx - n * K;
  float v = (k < Kb) ? base[(size_t)k * 128 + n] : spl[(size_t)(k - Kb) * 128 + n];
  out[idx] = f2bf(v);
}

// --- Signature level 1: S1 = w[S-1] - w[0] (telescoped prefix) --------------
__global__ void stk_sig1(const float* __restrict__ x, const float* __restrict__ tw,
                         float* __restrict__ sig) {
  int idx = blockIdx.x * blockDim.x + threadIdx.x;
  if (idx >= 64 * 64) return;
  int b = idx >> 6, d = idx & 63;
  float wl = tw[1023] * x[((size_t)b * 1024 + 1023) * 64 + d];
  float w0 = tw[0]    * x[((size_t)b * 1024) * 64 + d];
  sig[(size_t)b * 4160 + d] = wl - w0;
}

// --- Signature level 2: S2 = A^T U, per-batch 64x64 GEMM, K=1023 ------------
// a_t = 0.5*(w[t]+w[t+1]) - w[0],  u_t = w[t+1]-w[t].  One WG per batch,
// 8 waves x 2 tiles, K streamed through double-buffered LDS 32-step chunks.
__global__ __launch_bounds__(256) void stk_sig2(const float* __restrict__ x,
                                                const float* __restrict__ tw,
                                                float* __restrict__ sig) {
  int b = blockIdx.x;
  int tid = threadIdx.x;
  __shared__ __align__(16) __bf16 atT[2][64 * 32];   // [buf][d][t]
  __shared__ __align__(16) __bf16 utT[2][64 * 32];   // [buf][d][t]
  __shared__ float w0s[64];
  if (tid < 64) w0s[tid] = tw[0] * x[((size_t)b * 1024) * 64 + tid];

  int lane = tid & 31, wid = tid >> 5;
  int mt = wid >> 1;
  int nt0 = (wid & 1) * 2;
  int i  = mt * 16 + (lane & 15);
  int j0 = nt0 * 16 + (lane & 15);
  int j1 = (nt0 + 1) * 16 + (lane & 15);
  int kloA = (lane >> 4) << 4;
  int kbB  = (lane >> 4) << 4;
  f32x8 acc0 = zero8(), acc1 = zero8();

  auto fill = [&](int c) {
    __bf16* at = atT[c & 1];
    __bf16* ut = utT[c & 1];
    int t0 = c * 32;
#pragma unroll
    for (int it = 0; it < 8; ++it) {
      int idx = tid + it * 256;          // 2048 = 32 t x 64 d
      int tt = idx >> 6, d = idx & 63;
      int s0 = t0 + tt;
      float a = 0.f, u = 0.f;
      if (s0 < 1023) {
        float wa = tw[s0]     * x[((size_t)b * 1024 + s0) * 64 + d];
        float wb = tw[s0 + 1] * x[((size_t)b * 1024 + s0 + 1) * 64 + d];
        u = wb - wa;
        a = 0.5f * (wb + wa) - w0s[d];
      }
      at[d * 32 + tt] = f2bf(a);
      ut[d * 32 + tt] = f2bf(u);
    }
  };

  fill(0);
  __syncthreads();
  for (int c = 0; c < 32; ++c) {
    if (c + 1 < 32) fill(c + 1);           // writes buf (c+1)&1: disjoint
    const __bf16* at = atT[c & 1];
    const __bf16* ut = utT[c & 1];
    FragBF A, B0, B1;
    A.h[0]  = *(const bf16x8*)(&at[i * 32 + kloA]);
    A.h[1]  = *(const bf16x8*)(&at[i * 32 + kloA + 8]);
    B0.h[0] = *(const bf16x8*)(&ut[j0 * 32 + kbB]);
    B0.h[1] = *(const bf16x8*)(&ut[j0 * 32 + kbB + 8]);
    B1.h[0] = *(const bf16x8*)(&ut[j1 * 32 + kbB]);
    B1.h[1] = *(const bf16x8*)(&ut[j1 * 32 + kbB + 8]);
    acc0 = __builtin_amdgcn_wmma_f32_16x16x32_bf16(false, A.v, false, B0.v, (short)0, acc0, false, false);
    acc1 = __builtin_amdgcn_wmma_f32_16x16x32_bf16(false, A.v, false, B1.v, (short)0, acc1, false, false);
    __syncthreads();                       // one barrier per chunk
  }
  int M = (lane >> 4) << 3;
#pragma unroll
  for (int v = 0; v < 8; ++v) {
    int ii = mt * 16 + M + v;
    size_t base = (size_t)b * 4160 + 64 + (size_t)ii * 64;
    sig[base + nt0 * 16 + (lane & 15)]       = acc0[v];
    sig[base + (nt0 + 1) * 16 + (lane & 15)] = acc1[v];
  }
}

// --- KAN feature panel: X (R x F, f32) -> A (R x 9F, bf16) = [silu|bases] ---
__global__ void stk_panel(const float* __restrict__ X, __bf16* __restrict__ A,
                          int R, int F) {
  int idx = blockIdx.x * blockDim.x + threadIdx.x;
  if (idx >= R * F) return;
  int r = idx / F, f = idx - r * F;
  size_t K = (size_t)9 * F;
  float xv = X[(size_t)r * F + f];
  A[(size_t)r * K + f] = f2bf(siluf(xv));
  float bs[8];
  bspline8(xv, bs);
#pragma unroll
  for (int j = 0; j < 8; ++j)
    A[(size_t)r * K + F + (size_t)f * 8 + j] = f2bf(bs[j]);
}

// --- Split-K WMMA GEMM: A (M x K bf16) @ Wt^T (Wt 128 x K bf16) +=> C f32 ---
// grid = (M/16, 8, KSPLIT), block = 32. C must be zero-initialized.
__global__ __launch_bounds__(32) void stk_gemm(const __bf16* __restrict__ A,
                                               const __bf16* __restrict__ W,
                                               float* __restrict__ C,
                                               int K, int segChunks) {
  int mt = blockIdx.x, nt = blockIdx.y, ks = blockIdx.z;
  int lane = threadIdx.x;
  int chunks = K >> 5;
  int c0 = ks * segChunks;
  int c1 = c0 + segChunks; if (c1 > chunks) c1 = chunks;
  int m = mt * 16 + (lane & 15);
  int n = nt * 16 + (lane & 15);
  int kloA = (lane >> 4) << 4;
  int kbB  = (lane >> 4) << 4;
  const __bf16* arow = A + (size_t)m * K;
  const __bf16* wrow = W + (size_t)n * K;
  f32x8 acc = zero8();
  for (int c = c0; c < c1; ++c) {
    FragBF a, bfr;
    a.h[0]   = *(const bf16x8*)(arow + c * 32 + kloA);
    a.h[1]   = *(const bf16x8*)(arow + c * 32 + kloA + 8);
    bfr.h[0] = *(const bf16x8*)(wrow + c * 32 + kbB);
    bfr.h[1] = *(const bf16x8*)(wrow + c * 32 + kbB + 8);
    acc = __builtin_amdgcn_wmma_f32_16x16x32_bf16(false, a.v, false, bfr.v, (short)0, acc, false, false);
  }
  int row0 = mt * 16 + ((lane >> 4) << 3);
  int col  = nt * 16 + (lane & 15);
#pragma unroll
  for (int v = 0; v < 8; ++v)
    atomicAdd(&C[(size_t)(row0 + v) * 128 + col], acc[v]);
}

// --- skip = sig @ skip_w + skip_b (f32 exact; K=4160, split-K) --------------
__global__ __launch_bounds__(128) void stk_skip(const float* __restrict__ sig,
                                                const float* __restrict__ W,
                                                const float* __restrict__ bias,
                                                float* __restrict__ outp) {
  int b = blockIdx.x, ks = blockIdx.y, n = threadIdx.x;
  int k0 = ks * 520, k1 = k0 + 520;        // 8 * 520 = 4160
  const float* sr = sig + (size_t)b * 4160;
  float acc = (ks == 0) ? bias[n] : 0.f;
  for (int k = k0; k < k1; ++k)
    acc += sr[k] * W[(size_t)k * 128 + n];
  atomicAdd(&outp[b * 128 + n], acc);
}

// --- gate + layernorm + softmax -> per-batch mixing weights -----------------
__global__ __launch_bounds__(128) void stk_gate(const float* __restrict__ h2,
                                                const float* __restrict__ skip,
                                                const float* __restrict__ Ws, const float* __restrict__ bs,
                                                const float* __restrict__ Wv, const float* __restrict__ bv,
                                                const float* __restrict__ lng, const float* __restrict__ lnb,
                                                float* __restrict__ gw) {
  int b = blockIdx.x, n = threadIdx.x;
  __shared__ float hrow[128];
  __shared__ float red[128];
  hrow[n] = h2[b * 128 + n];
  __syncthreads();
  float sa = bs[n], va = bv[n];
  for (int k = 0; k < 128; ++k) {
    float hv = hrow[k];
    sa += hv * Ws[k * 128 + n];
    va += hv * Wv[k * 128 + n];
  }
  float g = (1.f / (1.f + __expf(-sa))) * va;
  float y = skip[b * 128 + n] + g;

  red[n] = y; __syncthreads();
  for (int s = 64; s > 0; s >>= 1) { if (n < s) red[n] += red[n + s]; __syncthreads(); }
  float mu = red[0] * (1.f / 128.f); __syncthreads();
  float dy = y - mu;
  red[n] = dy * dy; __syncthreads();
  for (int s = 64; s > 0; s >>= 1) { if (n < s) red[n] += red[n + s]; __syncthreads(); }
  float var = red[0] * (1.f / 128.f); __syncthreads();
  float z = lng[n] * dy * rsqrtf(var + 1e-3f) + lnb[n];

  red[n] = z; __syncthreads();
  for (int s = 64; s > 0; s >>= 1) { if (n < s) red[n] = fmaxf(red[n], red[n + s]); __syncthreads(); }
  float mx = red[0]; __syncthreads();
  float e = __expf(z - mx);
  red[n] = e; __syncthreads();
  for (int s = 64; s > 0; s >>= 1) { if (n < s) red[n] += red[n + s]; __syncthreads(); }
  gw[b * 128 + n] = e / red[0];
}

// --- Fused final KAN: 512 WGs x 128 rows --------------------------------
// x panel staged via GLOBAL_LOAD_ASYNC_TO_LDS_B128 (ASYNCcnt), A panel
// (silu | spline bases) built per 32-wide K-chunk in double-buffered LDS.
__global__ __launch_bounds__(256) void stk_final(const float* __restrict__ x,
                                                 const float* __restrict__ tw,
                                                 const __bf16* __restrict__ Wtk,
                                                 const float* __restrict__ gw,
                                                 float* __restrict__ out) {
  int wg = blockIdx.x;                       // 512 = 65536 rows / 128
  int tid = threadIdx.x;
  __shared__ __align__(16) float wloc[128 * 64];      // raw x panel, 32 KB
  __shared__ float tws[128];                           // tw per row
  __shared__ __align__(16) __bf16 Ac[2][128 * 32];     // double-buffered, 16 KB
  size_t row0 = (size_t)wg * 128;

  if (tid < 128) tws[tid] = tw[(int)((row0 + tid) & 1023)];
  // Async copy 128x64 f32 x-panel into LDS: 2048 lanes x 16B, no VGPR round-trip.
#pragma unroll
  for (int it = 0; it < 8; ++it) {
    int idx16 = tid + it * 256;              // 0..2047, 16B granules
    const float* gptr = x + (row0 + (idx16 >> 4)) * 64 + (idx16 & 15) * 4;
    unsigned lds = (unsigned)(unsigned long long)(&wloc[idx16 * 4]); // generic VA low 32 = LDS offset
    asm volatile("global_load_async_to_lds_b128 %0, %1, off"
                 :: "v"(lds), "v"(gptr) : "memory");
  }
  asm volatile("s_wait_asynccnt 0x0" ::: "memory");
  __syncthreads();

  int lane = tid & 31, wid = tid >> 5;
  int mrow = wid * 16 + (lane & 15);
  int kloA = (lane >> 4) << 4;
  int kbB  = (lane >> 4) << 4;
  f32x8 acc[8];
#pragma unroll
  for (int nt = 0; nt < 8; ++nt) acc[nt] = zero8();

  auto genChunk = [&](int c) {
    __bf16* buf = Ac[c & 1];
    if (c < 2) {
#pragma unroll
      for (int it = 0; it < 16; ++it) {
        int idx = tid + it * 256;            // 4096
        int r = idx >> 5, f = c * 32 + (idx & 31);
        buf[idx] = f2bf(siluf(wloc[r * 64 + f] * tws[r]));
      }
    } else {
      int fbase = (c - 2) * 4;
#pragma unroll
      for (int it = 0; it < 2; ++it) {
        int task = tid + it * 256;           // 512 = 128 rows x 4 feats
        int r = task & 127, fo = task >> 7;
        float v = wloc[r * 64 + fbase + fo] * tws[r];
        float bsv[8];
        bspline8(v, bsv);
#pragma unroll
        for (int jj = 0; jj < 8; ++jj)
          buf[r * 32 + fo * 8 + jj] = f2bf(bsv[jj]);
      }
    }
  };

  genChunk(0);
  __syncthreads();
  for (int c = 0; c < 18; ++c) {             // K = 64 silu + 512 bases = 576
    if (c + 1 < 18) genChunk(c + 1);         // writes the other buffer
    const __bf16* buf = Ac[c & 1];
    FragBF A;
    A.h[0] = *(const bf16x8*)(&buf[mrow * 32 + kloA]);
    A.h[1] = *(const bf16x8*)(&buf[mrow * 32 + kloA + 8]);
#pragma unroll
    for (int nt = 0; nt < 8; ++nt) {
      int n = nt * 16 + (lane & 15);
      const __bf16* wrow = Wtk + (size_t)n * 576 + c * 32 + kbB;
      FragBF Bf;
      Bf.h[0] = *(const bf16x8*)(wrow);
      Bf.h[1] = *(const bf16x8*)(wrow + 8);
      acc[nt] = __builtin_amdgcn_wmma_f32_16x16x32_bf16(false, A.v, false, Bf.v, (short)0, acc[nt], false, false);
    }
    __syncthreads();                         // one barrier per chunk
  }

  int bidx = wg >> 3;                        // 128 rows never cross a batch
  int M = (lane >> 4) << 3;
#pragma unroll
  for (int nt = 0; nt < 8; ++nt) {
    int n = nt * 16 + (lane & 15);
    float gwv = gw[bidx * 128 + n];
#pragma unroll
    for (int v = 0; v < 8; ++v) {
      size_t r = row0 + wid * 16 + M + v;
      out[r * 128 + n] = acc[nt][v] * gwv;
    }
  }
}

// ---------------------------------------------------------------------------
extern "C" void kernel_launch(void* const* d_in, const int* in_sizes, int n_in,
                              void* d_out, int out_size, void* d_ws, size_t ws_size,
                              hipStream_t stream) {
  const float* x        = (const float*)d_in[0];
  const float* tw       = (const float*)d_in[1];
  const float* kan_base = (const float*)d_in[2];
  const float* kan_spl  = (const float*)d_in[3];
  const float* g1_base  = (const float*)d_in[4];
  const float* g1_spl   = (const float*)d_in[5];
  const float* g2_base  = (const float*)d_in[6];
  const float* g2_spl   = (const float*)d_in[7];
  const float* skip_w   = (const float*)d_in[8];
  const float* skip_b   = (const float*)d_in[9];
  const float* gate_ws  = (const float*)d_in[10];
  const float* gate_bs  = (const float*)d_in[11];
  const float* gate_wv  = (const float*)d_in[12];
  const float* gate_bv  = (const float*)d_in[13];
  const float* ln_g     = (const float*)d_in[14];
  const float* ln_b     = (const float*)d_in[15];
  float* out = (float*)d_out;

  // workspace carve-out (~15.5 MB total)
  char* ws = (char*)d_ws;
  size_t off = 0;
  auto carve = [&](size_t bytes) -> char* {
    char* p = ws + off;
    off += (bytes + 255) & ~(size_t)255;
    return p;
  };
  float*  sig   = (float*)carve((size_t)64 * 4160 * 4);
  __bf16* A1    = (__bf16*)carve((size_t)64 * 37440 * 2);
  __bf16* Wt1   = (__bf16*)carve((size_t)128 * 37440 * 2);
  __bf16* A2    = (__bf16*)carve((size_t)64 * 1152 * 2);
  __bf16* Wt2   = (__bf16*)carve((size_t)128 * 1152 * 2);
  __bf16* Wtk   = (__bf16*)carve((size_t)128 * 576 * 2);
  float*  h1    = (float*)carve((size_t)64 * 128 * 4);
  float*  h2    = (float*)carve((size_t)64 * 128 * 4);
  float*  skipo = (float*)carve((size_t)64 * 128 * 4);
  float*  gw    = (float*)carve((size_t)64 * 128 * 4);

  // split-K targets must start at zero every launch
  hipMemsetAsync(h1,    0, (size_t)64 * 128 * 4, stream);
  hipMemsetAsync(h2,    0, (size_t)64 * 128 * 4, stream);
  hipMemsetAsync(skipo, 0, (size_t)64 * 128 * 4, stream);

  dim3 b256(256);
  // 1) bf16 transposed weights (one-time, L2 resident afterwards)
  stk_wconv<<<dim3((128 * 576   + 255) / 256), b256, 0, stream>>>(kan_base, kan_spl, Wtk, 64, 512);
  stk_wconv<<<dim3((128 * 37440 + 255) / 256), b256, 0, stream>>>(g1_base, g1_spl, Wt1, 4160, 33280);
  stk_wconv<<<dim3((128 * 1152  + 255) / 256), b256, 0, stream>>>(g2_base, g2_spl, Wt2, 128, 1024);
  // 2) signature (telescoped level-1 + WMMA level-2)
  stk_sig1<<<dim3(16), b256, 0, stream>>>(x, tw, sig);
  stk_sig2<<<dim3(64), b256, 0, stream>>>(x, tw, sig);
  // 3) GRKAN head (split-K WMMA GEMMs)
  stk_panel<<<dim3((64 * 4160 + 255) / 256), b256, 0, stream>>>(sig, A1, 64, 4160);
  stk_gemm<<<dim3(4, 8, 16), dim3(32), 0, stream>>>(A1, Wt1, h1, 37440, 74);  // 1170 chunks
  stk_panel<<<dim3((64 * 128 + 255) / 256), b256, 0, stream>>>(h1, A2, 64, 128);
  stk_gemm<<<dim3(4, 8, 4), dim3(32), 0, stream>>>(A2, Wt2, h2, 1152, 9);     // 36 chunks
  stk_skip<<<dim3(64, 8), dim3(128), 0, stream>>>(sig, skip_w, skip_b, skipo);
  stk_gate<<<dim3(64), dim3(128), 0, stream>>>(h2, skipo, gate_ws, gate_bs,
                                               gate_wv, gate_bv, ln_g, ln_b, gw);
  // 4) fused final KAN + gate scaling (the 9.7 GFLOP stage)
  stk_final<<<dim3(512), b256, 0, stream>>>(x, tw, Wtk, gw, out);

  (void)in_sizes; (void)n_in; (void)out_size; (void)ws_size;
}
